// BaseNonLocalBlock_74096775791099
// MI455X (gfx1250) — matrix operand order repaired and checked by reading, not compiled, not run on hardware
//
#include <hip/hip_runtime.h>
#include <hip/hip_bf16.h>

typedef __attribute__((ext_vector_type(16))) _Float16 v16h;
typedef __attribute__((ext_vector_type(8)))  float    v8f;

#define DCH   256
#define NPTS  4096
#define NHEAD 4
#define DHEAD 64

__device__ __forceinline__ v8f wmma16(v16h a, v16h b, v8f c) {
  // D = A(16x32 f16) * B(32x16 f16) + C(16x16 f32)
  return __builtin_amdgcn_wmma_f32_16x16x32_f16(false, a, false, b, (short)0, c,
                                                false, false);
}

// ---------------------------------------------------------------------------
// Generic WMMA GEMM:  D[M,N] = alpha * (A[M,K](f32) * B[K,N] + bias) + epilogue
//  MODE 0: B = f32 row-major [K][N] (x input), out f16 transposed [N][M]
//  MODE 1: B = f16 [N][K] (k-contiguous), out f16 transposed [N][M], BN+ReLU
//  MODE 2: B = f16 [N][K],                out f32 plain [M][N], +residual
//  MODE 3: B = f32 row-major [K][N],      out f16 plain [M][N]  (V matrix)
// Block = 256 threads = 8 waves arranged 4(M) x 2(N); WG tile 64x64,
// each wave computes a 16x32 strip (two 16x16 WMMA tiles), K-loop step 32.
// ---------------------------------------------------------------------------
template<int MODE>
__global__ __launch_bounds__(256) void gemm16(
    const float* __restrict__ A, const float* __restrict__ bias,
    const float* __restrict__ Bf32, const _Float16* __restrict__ Bf16,
    int K, int N, int ldB,
    _Float16* __restrict__ Out16, float* __restrict__ Out32, int M,
    const float* __restrict__ g, const float* __restrict__ be,
    const float* __restrict__ resid, float alpha)
{
  const int tid  = threadIdx.x;
  const int wave = tid >> 5, lane = tid & 31;
  const int hi = lane >> 4, ln = lane & 15;
  const int m0 = blockIdx.y * 64 + (wave >> 1) * 16;
  const int n0 = blockIdx.x * 64 + (wave & 1) * 32;

  v8f acc0 = {}, acc1 = {};
  for (int k0 = 0; k0 < K; k0 += 32) {
    // A fragment (16x32): lane<16 -> k {0..15}, lane>=16 -> k {16..31}
    v16h a;
    {
      const float* p = A + (size_t)(m0 + ln) * K + k0 + hi * 16;
      #pragma unroll
      for (int j = 0; j < 8; ++j) {
        a[j]     = (_Float16)p[j];
        a[8 + j] = (_Float16)p[8 + j];
      }
    }
    v16h b0, b1;
    if (MODE == 0 || MODE == 3) {
      const int kb = k0 + (hi << 4);
      #pragma unroll
      for (int e = 0; e < 16; ++e)
        b0[e] = (_Float16)Bf32[(size_t)(kb + e) * ldB + n0 + ln];
      #pragma unroll
      for (int e = 0; e < 16; ++e)
        b1[e] = (_Float16)Bf32[(size_t)(kb + e) * ldB + n0 + 16 + ln];
    } else {
      // B stored [N][K]: 32B contiguous per lane -> vector loads
      b0 = *(const v16h*)(Bf16 + (size_t)(n0 + ln)      * ldB + k0 + (hi << 4));
      b1 = *(const v16h*)(Bf16 + (size_t)(n0 + 16 + ln) * ldB + k0 + (hi << 4));
    }
    acc0 = wmma16(a, b0, acc0);
    acc1 = wmma16(a, b1, acc1);
  }

  #pragma unroll
  for (int r = 0; r < 8; ++r) {
    const int m = m0 + r + hi * 8;           // C/D layout: row = r + 8*half
    const float bv = bias[m];
    float v0 = (acc0[r] + bv) * alpha, v1 = (acc1[r] + bv) * alpha;
    if (MODE == 1) {
      const float s = g[m] * rsqrtf(1.0f + 1e-5f);
      const float bb = be[m];
      v0 = fmaxf(v0 * s + bb, 0.0f);
      v1 = fmaxf(v1 * s + bb, 0.0f);
    }
    if (MODE == 2) {
      const int na = n0 + ln, nb = n0 + 16 + ln;
      Out32[(size_t)m * N + na] = v0 + resid[(size_t)m * N + na];
      Out32[(size_t)m * N + nb] = v1 + resid[(size_t)m * N + nb];
    } else if (MODE == 3) {
      Out16[(size_t)m * N + n0 + ln]      = (_Float16)v0;
      Out16[(size_t)m * N + n0 + 16 + ln] = (_Float16)v1;
    } else {
      Out16[(size_t)(n0 + ln)      * M + m] = (_Float16)v0;
      Out16[(size_t)(n0 + 16 + ln) * M + m] = (_Float16)v1;
    }
  }
}

// ---------------------------------------------------------------------------
// Fused masked-softmax attention, no cross-lane shuffles.
//  Qt: [N][D] f16, pre-scaled by 1/sqrt(DH) (contiguous A fragments)
//  Kt: [N][D] f16 (contiguous B fragments for scores)
//  Vp: [D][N] f16 (contiguous B fragments for message)
//  sc: [N][N] f32 multiplicative mask; staged in LDS once per block and shared
//      by all 4 head-waves (sc is read exactly once from HBM in total)
//  mT: [N][D] f16 message transposed (feeds MLP GEMMs as k-contiguous B)
// Block = 128 threads = 4 waves, wave w == head w; each block owns 16 query
// rows. Keys stream in phases of 64 (two 32-key sub-tiles between barriers:
// 18 WMMAs per barrier phase). Softmax denominator is accumulated with a
// WMMA against an all-ones B matrix: every column of that D tile equals the
// row-sum of P, replicated across lanes in the accumulator register layout,
// so no cross-lane reductions are ever needed.
// ---------------------------------------------------------------------------
__global__ __launch_bounds__(128) void attn_flash(
    const _Float16* __restrict__ Qt, const _Float16* __restrict__ Kt,
    const _Float16* __restrict__ Vp, const float* __restrict__ sc,
    _Float16* __restrict__ mT)
{
  __shared__ float scT[16 * 64];
  __shared__ __align__(32) _Float16 Pa[NHEAD][2][32][16]; // P in A-frag layout

  const int tid  = threadIdx.x;
  const int wave = tid >> 5;            // head index
  const int lane = tid & 31;
  const int hi = lane >> 4, ln = lane & 15;
  const int o0 = blockIdx.x * 16;
  const int cbase = wave * DHEAD;

  // Q fragments for k=0..31 and k=32..63 (persist across the key loop)
  v16h aq0, aq1;
  {
    const _Float16* p = Qt + (size_t)(o0 + ln) * DCH + cbase + hi * 16;
    #pragma unroll
    for (int j = 0; j < 8; ++j) { aq0[j] = p[j];      aq0[8 + j] = p[8 + j]; }
    #pragma unroll
    for (int j = 0; j < 8; ++j) { aq1[j] = p[32 + j]; aq1[8 + j] = p[40 + j]; }
  }

  v16h ones;
  #pragma unroll
  for (int e = 0; e < 16; ++e) ones[e] = (_Float16)1.0f;

  const v8f zero = {};
  v8f acc0 = zero, acc1 = zero, acc2 = zero, acc3 = zero;
  v8f lacc = zero;                       // row-sums of exp(scores)

  for (int i0 = 0; i0 < NPTS; i0 += 64) {
    // stage sc tile [16 o][64 i] (protect previous readers, then publish)
    __syncthreads();
    #pragma unroll
    for (int q = 0; q < 8; ++q) {
      const int e = tid + q * 128;
      scT[e] = sc[(size_t)(o0 + (e >> 6)) * NPTS + i0 + (e & 63)];
    }
    __syncthreads();

    #pragma unroll
    for (int half = 0; half < 2; ++half) {
      const int ib = i0 + half * 32;     // base of this 32-key sub-tile
      const int cc = half * 32;          // column offset in scT

      // ---- scores: S = (Q_h/8)^T * K_h over c=0..63 (2 WMMA per tile) ----
      v16h bka = *(const v16h*)(Kt + (size_t)(ib + ln)      * DCH + cbase +      (hi << 4));
      v16h bkb = *(const v16h*)(Kt + (size_t)(ib + ln)      * DCH + cbase + 32 + (hi << 4));
      v8f s0 = wmma16(aq0, bka, zero);
      s0     = wmma16(aq1, bkb, s0);
      v16h bkc = *(const v16h*)(Kt + (size_t)(ib + 16 + ln) * DCH + cbase +      (hi << 4));
      v16h bkd = *(const v16h*)(Kt + (size_t)(ib + 16 + ln) * DCH + cbase + 32 + (hi << 4));
      v8f s1 = wmma16(aq0, bkc, zero);
      s1     = wmma16(aq1, bkd, s1);

      // ---- P = exp(sc * att); scatter into the 16x32 A-fragment layout ----
      #pragma unroll
      for (int r = 0; r < 8; ++r) {
        const int row = r + 8 * hi;      // C/D layout: row = r + 8*half-wave
        const float x0 = __expf(s0[r] * scT[row * 64 + cc + ln]);
        const float x1 = __expf(s1[r] * scT[row * 64 + cc + 16 + ln]);
        const int la0 = row;                           // cols {0..15}
        Pa[wave][half][la0][ln] = (_Float16)x0;
        const int la1 = row + 16;                      // cols {16..31}
        const int e1  = ln;
        Pa[wave][half][la1][e1] = (_Float16)x1;
      }

      v16h ap = *(const v16h*)&Pa[wave][half][lane][0]; // per-lane contiguous

      // softmax denominator via ones-WMMA (row-sum replicated across lanes)
      lacc = wmma16(ap, ones, lacc);

      // ---- message: acc += P(16x32) * V_h^T(32x64), 4 WMMA ----
      const int kb = ib + (hi << 4);
      const _Float16* vb = Vp + (size_t)cbase * NPTS + kb;
      v16h bv0 = *(const v16h*)(vb + (size_t)(ln)      * NPTS);
      acc0 = wmma16(ap, bv0, acc0);
      v16h bv1 = *(const v16h*)(vb + (size_t)(16 + ln) * NPTS);
      acc1 = wmma16(ap, bv1, acc1);
      v16h bv2 = *(const v16h*)(vb + (size_t)(32 + ln) * NPTS);
      acc2 = wmma16(ap, bv2, acc2);
      v16h bv3 = *(const v16h*)(vb + (size_t)(48 + ln) * NPTS);
      acc3 = wmma16(ap, bv3, acc3);
    }
  }

  // normalize by softmax denominator and emit message^T [o][c]
  #pragma unroll
  for (int r = 0; r < 8; ++r) {
    const float inv = 1.0f / lacc[r];
    const int o = o0 + r + 8 * hi;
    _Float16* dst = mT + (size_t)o * DCH + cbase;
    dst[0 * 16 + ln] = (_Float16)(acc0[r] * inv);
    dst[1 * 16 + ln] = (_Float16)(acc1[r] * inv);
    dst[2 * 16 + ln] = (_Float16)(acc2[r] * inv);
    dst[3 * 16 + ln] = (_Float16)(acc3[r] * inv);
  }
}

// ---------------------------------------------------------------------------
extern "C" void kernel_launch(void* const* d_in, const int* in_sizes, int n_in,
                              void* d_out, int out_size, void* d_ws, size_t ws_size,
                              hipStream_t stream) {
  (void)in_sizes; (void)n_in; (void)out_size; (void)ws_size;
  const float* x   = (const float*)d_in[0];   // [256][4096]
  const float* sc  = (const float*)d_in[1];   // [4096][4096]
  const float* Wq  = (const float*)d_in[2];
  const float* bq  = (const float*)d_in[3];
  const float* Wk  = (const float*)d_in[4];
  const float* bk  = (const float*)d_in[5];
  const float* Wv  = (const float*)d_in[6];
  const float* bv  = (const float*)d_in[7];
  const float* W1  = (const float*)d_in[8];
  const float* b1  = (const float*)d_in[9];
  const float* g1  = (const float*)d_in[10];
  const float* be1 = (const float*)d_in[11];
  const float* W2  = (const float*)d_in[12];
  const float* b2  = (const float*)d_in[13];
  const float* g2  = (const float*)d_in[14];
  const float* be2 = (const float*)d_in[15];
  const float* W3  = (const float*)d_in[16];
  const float* b3  = (const float*)d_in[17];
  float* out = (float*)d_out;

  char* ws = (char*)d_ws;
  _Float16* Qt  = (_Float16*)(ws + ((size_t)0 << 20));  // [4096][256] 2MB
  _Float16* Kt  = (_Float16*)(ws + ((size_t)2 << 20));  // [4096][256] 2MB
  _Float16* Vp  = (_Float16*)(ws + ((size_t)4 << 20));  // [256][4096] 2MB
  _Float16* mT  = (_Float16*)(ws + ((size_t)6 << 20));  // [4096][256] 2MB
  _Float16* h1T = (_Float16*)(ws + ((size_t)8 << 20));  // [4096][128] 1MB
  _Float16* h2T = (_Float16*)(ws + ((size_t)9 << 20));  // [4096][128] 1MB

  const dim3 blk(256);
  // Q pre-scaled by 1/sqrt(DH)=0.125; K transposed; V stored plain [D][N]
  gemm16<0><<<dim3(64, 4), blk, 0, stream>>>(Wq, bq, x, nullptr, 256, 4096, 4096,
                                             Qt, nullptr, 256, nullptr, nullptr,
                                             nullptr, 0.125f);
  gemm16<0><<<dim3(64, 4), blk, 0, stream>>>(Wk, bk, x, nullptr, 256, 4096, 4096,
                                             Kt, nullptr, 256, nullptr, nullptr,
                                             nullptr, 1.0f);
  gemm16<3><<<dim3(64, 4), blk, 0, stream>>>(Wv, bv, x, nullptr, 256, 4096, 4096,
                                             Vp, nullptr, 256, nullptr, nullptr,
                                             nullptr, 1.0f);
  // fused masked-softmax attention, all 4 heads per block
  attn_flash<<<dim3(NPTS / 16), dim3(128), 0, stream>>>(Qt, Kt, Vp, sc, mT);
  // MLP: conv1x1 -> BN -> ReLU -> conv1x1 -> BN -> ReLU -> conv1x1 + residual
  gemm16<1><<<dim3(64, 2), blk, 0, stream>>>(W1, b1, nullptr, mT, 256, 4096, 256,
                                             h1T, nullptr, 128, g1, be1,
                                             nullptr, 1.0f);
  gemm16<1><<<dim3(64, 2), blk, 0, stream>>>(W2, b2, nullptr, h1T, 128, 4096, 128,
                                             h2T, nullptr, 128, g2, be2,
                                             nullptr, 1.0f);
  gemm16<2><<<dim3(64, 4), blk, 0, stream>>>(W3, b3, nullptr, h2T, 128, 4096, 128,
                                             nullptr, out, 256, nullptr, nullptr,
                                             x, 1.0f);
}
